// SelfAttention_4647154614418
// MI455X (gfx1250) — matrix operand order repaired and checked by reading, not compiled, not run on hardware
//
#include <hip/hip_runtime.h>
#include <cstdint>

typedef __attribute__((ext_vector_type(16))) __bf16 v16bf;
typedef __attribute__((ext_vector_type(8)))  __bf16 v8bf;
typedef __attribute__((ext_vector_type(4)))  __bf16 v4bf;
typedef __attribute__((ext_vector_type(8)))  float  v8f;
typedef __attribute__((ext_vector_type(4)))  float  v4f;

#define BD 2
#define HD 16
#define SD 2048
#define DHD 64
#define KT 64            // keys per iteration
#define WAVES 8
#define QW 16            // query rows per wave
#define QTILE (WAVES*QW) // 128 query rows per block

// ---------------------------------------------------------------------------
// CDNA5 async-to-LDS copy (raw 16B per lane), tracked with ASYNCcnt.
// ---------------------------------------------------------------------------
__device__ __forceinline__ void async_b128(uint32_t lds_addr, uint64_t gaddr) {
    asm volatile("global_load_async_to_lds_b128 %0, %1, off"
                 :: "v"(lds_addr), "v"(gaddr) : "memory");
}

// ===========================================================================
// Pre-pass: K -> bf16 row-major [bh][key][dh]; V -> bf16 transposed
// [bh][dh][key] (per-head row stride SD), so the main loop's B-fragments are
// two 8-element runs in LDS with zero conversion work.
// ===========================================================================
__global__ __launch_bounds__(256) void convert_kv_kernel(
    const float* __restrict__ k,
    const float* __restrict__ v,
    __bf16* __restrict__ kbf,
    __bf16* __restrict__ vtb)
{
    __shared__ __align__(16) __bf16 T[DHD][KT];   // 8 KB transpose tile

    const int tiles = SD / KT;                    // 32
    const int bh = blockIdx.x / tiles;
    const int k0 = (blockIdx.x % tiles) * KT;
    const int tid = threadIdx.x;

    const float* kp = k + ((size_t)bh * SD + k0) * DHD;
    const float* vp = v + ((size_t)bh * SD + k0) * DHD;
    __bf16* ko = kbf + ((size_t)bh * SD + k0) * DHD;
    __bf16* vo = vtb + (size_t)bh * DHD * SD + k0;

    #pragma unroll
    for (int i = 0; i < 4; ++i) {
        const int f   = tid + 256 * i;            // 1024 float4s (64x64 f32)
        const int key = f >> 4;
        const int dc  = (f & 15) << 2;
        v4f kx = *(const v4f*)(kp + (size_t)key * DHD + dc);
        v4bf kb;
        #pragma unroll
        for (int t = 0; t < 4; ++t) kb[t] = (__bf16)kx[t];
        *(v4bf*)(ko + (size_t)key * DHD + dc) = kb;

        v4f vx = *(const v4f*)(vp + (size_t)key * DHD + dc);
        #pragma unroll
        for (int t = 0; t < 4; ++t) T[dc + t][key] = (__bf16)vx[t];
    }
    __syncthreads();
    #pragma unroll
    for (int i = 0; i < 2; ++i) {
        const int c   = tid + 256 * i;            // 512 chunks of 8 bf16
        const int dh  = c >> 3;
        const int kof = (c & 7) << 3;
        *(v8bf*)(vo + (size_t)dh * SD + kof) = *(const v8bf*)&T[dh][kof];
    }
}

// ===========================================================================
// Main flash-attention kernel: bf16 tiles double-buffered into LDS with
// GLOBAL_LOAD_ASYNC_TO_LDS_B128 (ASYNCcnt), WMMA bf16 with f32 accumulate.
// ===========================================================================
__global__ __launch_bounds__(256) void fa_fwd_async_kernel(
    const float* __restrict__ q,
    const __bf16* __restrict__ kbf,   // [bh][key][dh]
    const __bf16* __restrict__ vtb,   // [bh][dh][key]
    const uint8_t* __restrict__ kpm,
    float* __restrict__ out)
{
    __shared__ __align__(16) __bf16 Ks[2][KT * DHD];   // 2 x 8 KB  [key][dh]
    __shared__ __align__(16) __bf16 Vt[2][DHD * KT];   // 2 x 8 KB  [dh][key]
    __shared__ __align__(16) __bf16 Ps[WAVES][QW][KT]; // 16 KB P staging

    const int tid  = threadIdx.x;
    const int wave = tid >> 5;
    const int lane = tid & 31;
    const int half = lane >> 4;
    const int l16  = lane & 15;

    const int qtiles = SD / QTILE;        // 16
    const int bh = blockIdx.x / qtiles;
    const int qt = blockIdx.x % qtiles;
    const int b  = bh / HD;
    const int h  = bh % HD;
    const int q0 = qt * QTILE + wave * QW;

    const float* qp = q + (size_t)bh * SD * DHD;
    const __bf16* kh = kbf + (size_t)bh * SD * DHD;
    const __bf16* vh = vtb + (size_t)bh * DHD * SD;
    const uint8_t* mp = kpm + (size_t)b * SD;

    const float scale = 0.125f; // 1/sqrt(64)

    // ---- Q A-fragments (16x32 bf16 each, dh halves), scale folded in ----
    v16bf qa[2];
    {
        const int row = q0 + l16;
        const float* qrow = qp + (size_t)row * DHD;
        #pragma unroll
        for (int j = 0; j < 2; ++j) {
            v16bf a;
            #pragma unroll
            for (int run = 0; run < 2; ++run) {
                const int d0 = 32*j + 16*run + 8*half;
                v4f x0 = *(const v4f*)(qrow + d0);
                v4f x1 = *(const v4f*)(qrow + d0 + 4);
                #pragma unroll
                for (int t = 0; t < 4; ++t) {
                    a[8*run + t]     = (__bf16)(x0[t] * scale);
                    a[8*run + 4 + t] = (__bf16)(x1[t] * scale);
                }
            }
            qa[j] = a;
        }
    }

    // ---- accumulators ----
    v8f Oacc[4];
    float mrow[8], lrow[8];
    #pragma unroll
    for (int d = 0; d < 4; ++d)
        #pragma unroll
        for (int r = 0; r < 8; ++r) Oacc[d][r] = 0.0f;
    #pragma unroll
    for (int r = 0; r < 8; ++r) { mrow[r] = -__builtin_inff(); lrow[r] = 0.0f; }

    // async tile issue: 4 async instructions per wave per tile
    auto issue_tile = [&](int k0, int buf) {
        const uint64_t gk = (uint64_t)(uintptr_t)(kh + (size_t)k0 * DHD);
        const uint32_t lk = (uint32_t)(uintptr_t)(&Ks[buf][0]);
        async_b128(lk + tid * 16,        gk + tid * 16);         // 8 KB contiguous
        async_b128(lk + tid * 16 + 4096, gk + tid * 16 + 4096);
        const uint64_t gv = (uint64_t)(uintptr_t)(vh + (size_t)k0);
        const uint32_t lv = (uint32_t)(uintptr_t)(&Vt[buf][0]);
        const int c0 = tid, c1 = tid + 256;                      // 512 x 16B rows-of-128B
        async_b128(lv + c0 * 16, gv + (uint64_t)(c0 >> 3) * (SD * 2) + (c0 & 7) * 16);
        async_b128(lv + c1 * 16, gv + (uint64_t)(c1 >> 3) * (SD * 2) + (c1 & 7) * 16);
    };

    const int nt = SD / KT;   // 32
    issue_tile(0, 0);

    for (int it = 0; it < nt; ++it) {
        const int buf = it & 1;
        const int k0 = it * KT;

        if (it + 1 < nt) {
            issue_tile(k0 + KT, buf ^ 1);
            asm volatile("s_wait_asynccnt 0x4" ::: "memory"); // current tile done
        } else {
            asm volatile("s_wait_asynccnt 0x0" ::: "memory");
        }
        __syncthreads();   // current tile visible to all waves

        // ---- scores S = (Q*scale) @ K^T ----
        v8f sc[4];
        #pragma unroll
        for (int n = 0; n < 4; ++n) {
            v8f c;
            #pragma unroll
            for (int r = 0; r < 8; ++r) c[r] = 0.0f;
            #pragma unroll
            for (int j = 0; j < 2; ++j) {
                const __bf16* src = &Ks[buf][(16*n + l16) * DHD + 32*j + 8*half];
                v8bf c0 = *(const v8bf*)src;
                v8bf c1 = *(const v8bf*)(src + 16);
                v16bf bfrag = __builtin_shufflevector(c0, c1,
                    0,1,2,3,4,5,6,7,8,9,10,11,12,13,14,15);
                c = __builtin_amdgcn_wmma_f32_16x16x32_bf16(
                        false, qa[j], false, bfrag, (short)0, c, false, false);
            }
            const bool msk = mp[k0 + 16*n + l16] != 0;
            #pragma unroll
            for (int r = 0; r < 8; ++r)
                c[r] = msk ? -__builtin_inff() : c[r];
            sc[n] = c;
        }

        // ---- online softmax ----
        float mnew[8], alpha[8];
        #pragma unroll
        for (int r = 0; r < 8; ++r) {
            float mx = fmaxf(fmaxf(sc[0][r], sc[1][r]), fmaxf(sc[2][r], sc[3][r]));
            #pragma unroll
            for (int off = 1; off < 16; off <<= 1)
                mx = fmaxf(mx, __shfl_xor(mx, off, 32));
            mnew[r]  = fmaxf(mrow[r], mx);
            alpha[r] = (mrow[r] <= -1e37f) ? 0.0f : __expf(mrow[r] - mnew[r]);
            mrow[r]  = mnew[r];
        }
        #pragma unroll
        for (int r = 0; r < 8; ++r) {
            float s = 0.0f;
            #pragma unroll
            for (int n = 0; n < 4; ++n) {
                float p = __expf(sc[n][r] - mnew[r]);
                sc[n][r] = p;
                s += p;
            }
            #pragma unroll
            for (int off = 1; off < 16; off <<= 1)
                s += __shfl_xor(s, off, 32);
            lrow[r] = lrow[r] * alpha[r] + s;
        }

        #pragma unroll
        for (int d = 0; d < 4; ++d)
            #pragma unroll
            for (int r = 0; r < 8; ++r) Oacc[d][r] *= alpha[r];

        // ---- C-layout P -> A-layout via per-wave LDS staging ----
        #pragma unroll
        for (int n = 0; n < 4; ++n)
            #pragma unroll
            for (int r = 0; r < 8; ++r)
                Ps[wave][r + 8*half][16*n + l16] = (__bf16)sc[n][r];

        asm volatile("s_wait_dscnt 0" ::: "memory");

        // ---- O += P @ V ----
        #pragma unroll
        for (int j = 0; j < 2; ++j) {
            const __bf16* prow = &Ps[wave][l16][32*j];
            v8bf p0 = *(const v8bf*)(prow + 8*half);
            v8bf p1 = *(const v8bf*)(prow + 16 + 8*half);
            v16bf pa = __builtin_shufflevector(p0, p1,
                0,1,2,3,4,5,6,7,8,9,10,11,12,13,14,15);
            #pragma unroll
            for (int d = 0; d < 4; ++d) {
                const __bf16* vsrc = &Vt[buf][(16*d + l16) * KT + 32*j + 8*half];
                v8bf b0 = *(const v8bf*)vsrc;
                v8bf b1 = *(const v8bf*)(vsrc + 16);
                v16bf vb = __builtin_shufflevector(b0, b1,
                    0,1,2,3,4,5,6,7,8,9,10,11,12,13,14,15);
                Oacc[d] = __builtin_amdgcn_wmma_f32_16x16x32_bf16(
                              false, pa, false, vb, (short)0, Oacc[d], false, false);
            }
        }
        __syncthreads();   // done reading buf before it is overwritten
    }

    // ---- finalize & store to [B, S, H*DH] ----
    #pragma unroll
    for (int r = 0; r < 8; ++r) {
        const float inv = 1.0f / lrow[r];
        const int row = q0 + r + 8*half;
        float* orow = out + ((size_t)b*SD + row)*(HD*DHD) + h*DHD;
        #pragma unroll
        for (int d = 0; d < 4; ++d)
            orow[16*d + l16] = Oacc[d][r] * inv;
    }
}

// ===========================================================================
// Fallback: round-1 fused kernel (inline f32->bf16 conversion), used only if
// the workspace cannot hold the pre-converted bf16 K / V-transposed tensors.
// ===========================================================================
__global__ __launch_bounds__(256) void fa_fwd_fused_kernel(
    const float* __restrict__ q,
    const float* __restrict__ k,
    const float* __restrict__ v,
    const uint8_t* __restrict__ kpm,
    float* __restrict__ out)
{
    __shared__ __align__(16) __bf16 Ks[KT][DHD];
    __shared__ __align__(16) __bf16 Vt[DHD][KT];
    __shared__ __align__(16) __bf16 Ps[WAVES][QW][KT];

    const int tid  = threadIdx.x;
    const int wave = tid >> 5;
    const int lane = tid & 31;
    const int half = lane >> 4;
    const int l16  = lane & 15;

    const int qtiles = SD / QTILE;
    const int bh = blockIdx.x / qtiles;
    const int qt = blockIdx.x % qtiles;
    const int b  = bh / HD;
    const int h  = bh % HD;
    const int q0 = qt * QTILE + wave * QW;

    const size_t headOff = (size_t)bh * SD * DHD;
    const float* qp = q + headOff;
    const float* kp = k + headOff;
    const float* vp = v + headOff;
    const uint8_t* mp = kpm + (size_t)b * SD;

    const float scale = 0.125f;

    v16bf qa[2];
    {
        const int row = q0 + l16;
        const float* qrow = qp + (size_t)row * DHD;
        #pragma unroll
        for (int j = 0; j < 2; ++j) {
            v16bf a;
            #pragma unroll
            for (int run = 0; run < 2; ++run) {
                const int d0 = 32*j + 16*run + 8*half;
                v4f x0 = *(const v4f*)(qrow + d0);
                v4f x1 = *(const v4f*)(qrow + d0 + 4);
                #pragma unroll
                for (int t = 0; t < 4; ++t) {
                    a[8*run + t]     = (__bf16)(x0[t] * scale);
                    a[8*run + 4 + t] = (__bf16)(x1[t] * scale);
                }
            }
            qa[j] = a;
        }
    }

    v8f Oacc[4];
    float mrow[8], lrow[8];
    #pragma unroll
    for (int d = 0; d < 4; ++d)
        #pragma unroll
        for (int r = 0; r < 8; ++r) Oacc[d][r] = 0.0f;
    #pragma unroll
    for (int r = 0; r < 8; ++r) { mrow[r] = -__builtin_inff(); lrow[r] = 0.0f; }

    for (int k0 = 0; k0 < SD; k0 += KT) {
        #pragma unroll
        for (int i = 0; i < 4; ++i) {
            const int f   = tid + 256*i;
            const int key = f >> 4;
            const int dc  = (f & 15) << 2;
            v4f kx = *(const v4f*)(kp + (size_t)(k0 + key)*DHD + dc);
            v4bf kb;
            #pragma unroll
            for (int t = 0; t < 4; ++t) kb[t] = (__bf16)kx[t];
            *(v4bf*)&Ks[key][dc] = kb;
            v4f vx = *(const v4f*)(vp + (size_t)(k0 + key)*DHD + dc);
            #pragma unroll
            for (int t = 0; t < 4; ++t) Vt[dc + t][key] = (__bf16)vx[t];
        }
        __syncthreads();

        v8f sc[4];
        #pragma unroll
        for (int n = 0; n < 4; ++n) {
            v8f c;
            #pragma unroll
            for (int r = 0; r < 8; ++r) c[r] = 0.0f;
            #pragma unroll
            for (int j = 0; j < 2; ++j) {
                const __bf16* src = &Ks[16*n + l16][32*j + 8*half];
                v8bf c0 = *(const v8bf*)src;
                v8bf c1 = *(const v8bf*)(src + 16);
                v16bf bfrag = __builtin_shufflevector(c0, c1,
                    0,1,2,3,4,5,6,7,8,9,10,11,12,13,14,15);
                c = __builtin_amdgcn_wmma_f32_16x16x32_bf16(
                        false, qa[j], false, bfrag, (short)0, c, false, false);
            }
            const bool msk = mp[k0 + 16*n + l16] != 0;
            #pragma unroll
            for (int r = 0; r < 8; ++r)
                c[r] = msk ? -__builtin_inff() : c[r];
            sc[n] = c;
        }

        float mnew[8], alpha[8];
        #pragma unroll
        for (int r = 0; r < 8; ++r) {
            float mx = fmaxf(fmaxf(sc[0][r], sc[1][r]), fmaxf(sc[2][r], sc[3][r]));
            #pragma unroll
            for (int off = 1; off < 16; off <<= 1)
                mx = fmaxf(mx, __shfl_xor(mx, off, 32));
            mnew[r]  = fmaxf(mrow[r], mx);
            alpha[r] = (mrow[r] <= -1e37f) ? 0.0f : __expf(mrow[r] - mnew[r]);
            mrow[r]  = mnew[r];
        }
        #pragma unroll
        for (int r = 0; r < 8; ++r) {
            float s = 0.0f;
            #pragma unroll
            for (int n = 0; n < 4; ++n) {
                float p = __expf(sc[n][r] - mnew[r]);
                sc[n][r] = p;
                s += p;
            }
            #pragma unroll
            for (int off = 1; off < 16; off <<= 1)
                s += __shfl_xor(s, off, 32);
            lrow[r] = lrow[r] * alpha[r] + s;
        }

        #pragma unroll
        for (int d = 0; d < 4; ++d)
            #pragma unroll
            for (int r = 0; r < 8; ++r) Oacc[d][r] *= alpha[r];

        #pragma unroll
        for (int n = 0; n < 4; ++n)
            #pragma unroll
            for (int r = 0; r < 8; ++r)
                Ps[wave][r + 8*half][16*n + l16] = (__bf16)sc[n][r];

        asm volatile("s_wait_dscnt 0" ::: "memory");

        #pragma unroll
        for (int j = 0; j < 2; ++j) {
            const __bf16* prow = &Ps[wave][l16][32*j];
            v8bf p0 = *(const v8bf*)(prow + 8*half);
            v8bf p1 = *(const v8bf*)(prow + 16 + 8*half);
            v16bf pa = __builtin_shufflevector(p0, p1,
                0,1,2,3,4,5,6,7,8,9,10,11,12,13,14,15);
            #pragma unroll
            for (int d = 0; d < 4; ++d) {
                const __bf16* vsrc = &Vt[16*d + l16][32*j + 8*half];
                v8bf b0 = *(const v8bf*)vsrc;
                v8bf b1 = *(const v8bf*)(vsrc + 16);
                v16bf vb = __builtin_shufflevector(b0, b1,
                    0,1,2,3,4,5,6,7,8,9,10,11,12,13,14,15);
                Oacc[d] = __builtin_amdgcn_wmma_f32_16x16x32_bf16(
                              false, pa, false, vb, (short)0, Oacc[d], false, false);
            }
        }
        __syncthreads();
    }

    #pragma unroll
    for (int r = 0; r < 8; ++r) {
        const float inv = 1.0f / lrow[r];
        const int row = q0 + r + 8*half;
        float* orow = out + ((size_t)b*SD + row)*(HD*DHD) + h*DHD;
        #pragma unroll
        for (int d = 0; d < 4; ++d)
            orow[16*d + l16] = Oacc[d][r] * inv;
    }
}

// ===========================================================================
extern "C" void kernel_launch(void* const* d_in, const int* in_sizes, int n_in,
                              void* d_out, int out_size, void* d_ws, size_t ws_size,
                              hipStream_t stream) {
    const float*   q   = (const float*)d_in[0];
    const float*   k   = (const float*)d_in[1];
    const float*   v   = (const float*)d_in[2];
    const uint8_t* msk = (const uint8_t*)d_in[3];
    float* out = (float*)d_out;

    const size_t elems = (size_t)BD * HD * SD * DHD;      // 4,194,304
    const size_t need  = 2 * elems * sizeof(__bf16);      // ~16.8 MB

    if (ws_size >= need) {
        __bf16* kbf = (__bf16*)d_ws;
        __bf16* vtb = kbf + elems;
        convert_kv_kernel<<<dim3(BD * HD * (SD / KT)), dim3(256), 0, stream>>>(
            k, v, kbf, vtb);
        fa_fwd_async_kernel<<<dim3(BD * HD * (SD / QTILE)), dim3(256), 0, stream>>>(
            q, kbf, vtb, msk, out);
    } else {
        fa_fwd_fused_kernel<<<dim3(BD * HD * (SD / QTILE)), dim3(256), 0, stream>>>(
            q, k, v, msk, out);
    }
}
